// PraxisAttention_38646115729875
// MI455X (gfx1250) — matrix operand order repaired and checked by reading, not compiled, not run on hardware
//
#include <hip/hip_runtime.h>
#include <hip/hip_bf16.h>

// ---------------------------------------------------------------------------
// Praxis linear-attention forward for MI455X (gfx1250, wave32, WMMA)
//   B=4, L=4096, D=2048, H=16, Dh=128
//   4x GEMM (16384x2048x2048) on v_wmma_f32_16x16x32_bf16 (fp32 accum),
//   double-buffered async global->LDS staging (ASYNCcnt),
//   chunked cumsum scan (wave32 shuffles), fused elu+1 / mask epilogues
// ---------------------------------------------------------------------------

typedef __bf16 bf16;
typedef __attribute__((ext_vector_type(16))) __bf16 v16bf;
typedef __attribute__((ext_vector_type(8)))  float  v8f;

#define DIM_B 4
#define DIM_L 4096
#define DIM_D 2048
#define DIM_H 16
#define DIM_DH 128
#define MTOK (DIM_B * DIM_L)        // 16384 rows
#define NCHUNK 64
#define CHUNK (DIM_L / NCHUNK)      // 64 tokens per chunk

// ---- CDNA5 async copy + wait helpers --------------------------------------
__device__ __forceinline__ void async_load_b128_to_lds(unsigned lds_byte_off,
                                                       const void* gaddr) {
    // GLOBAL_LOAD_ASYNC_TO_LDS_B128 (GV mode): per-lane 16B memory->LDS DMA.
    // VDST VGPR holds the LDS byte address; completion tracked by ASYNCcnt.
    asm volatile("global_load_async_to_lds_b128 %0, %1, off"
                 :: "v"(lds_byte_off), "v"(gaddr) : "memory");
}
__device__ __forceinline__ void wait_asynccnt_le8() {
#if __has_builtin(__builtin_amdgcn_s_wait_asynccnt)
    __builtin_amdgcn_s_wait_asynccnt(8);
#else
    asm volatile("s_wait_asynccnt 0x8" ::: "memory");
#endif
}
__device__ __forceinline__ void wait_asynccnt_0() {
#if __has_builtin(__builtin_amdgcn_s_wait_asynccnt)
    __builtin_amdgcn_s_wait_asynccnt(0);
#else
    asm volatile("s_wait_asynccnt 0x0" ::: "memory");
#endif
}

// ---------------- fp32 -> bf16 convert (grid-stride) ----------------
__global__ void cvt_f32_bf16(const float* __restrict__ in, bf16* __restrict__ out, int n) {
    for (int i = blockIdx.x * blockDim.x + threadIdx.x; i < n; i += gridDim.x * blockDim.x)
        out[i] = (bf16)in[i];
}

// ------------- fp32 W[k][n] -> bf16 Wt[n][k] (tiled transpose) -------------
__global__ void transpose_cvt(const float* __restrict__ W, bf16* __restrict__ Wt, int Dm) {
    __shared__ float tile[32][33];
    const int n0 = blockIdx.x * 32;
    const int k0 = blockIdx.y * 32;
    const int tx = threadIdx.x & 31;
    const int ty = threadIdx.x >> 5;      // 0..7
#pragma unroll
    for (int i = 0; i < 4; ++i)
        tile[ty + i * 8][tx] = W[(size_t)(k0 + ty + i * 8) * Dm + n0 + tx];
    __syncthreads();
#pragma unroll
    for (int i = 0; i < 4; ++i)
        Wt[(size_t)(n0 + ty + i * 8) * Dm + k0 + tx] = (bf16)tile[tx][ty + i * 8];
}

// ---------------- WMMA GEMM: C[M,N] = A[M,K] @ Bt[N,K]^T + bias -----------
// Block tile 128x128, BK=64 (two wmma k-steps per LDS stage), 8 waves:
// 2x4 wave grid, each wave owns a 64x32 C tile = 4x2 16x16 fragments.
// Double-buffered LDS filled by async global->LDS DMA (stage k+1 in flight
// while stage k feeds 16 WMMAs). EPI: 0=bias, 1=+elu1, 2=+elu1*mask, 3=*mask
#define LDSTRIDE 80   // 64 + 16 pad bf16 (160B rows: 16B aligned, bank-shifted)

template <int EPI>
__global__ __launch_bounds__(256) void gemm_bf16(
    const bf16* __restrict__ A, const bf16* __restrict__ Bt,
    const float* __restrict__ bias, const float* __restrict__ rowmask,
    float* __restrict__ out, int M, int N, int K) {
    __shared__ bf16 As[2][128 * LDSTRIDE];
    __shared__ bf16 Bs[2][128 * LDSTRIDE];

    const int tid  = threadIdx.x;
    const int lane = tid & 31;
    const int wave = tid >> 5;
    const int wm = wave & 1;        // 0..1 -> 64-row strip
    const int wn = wave >> 1;       // 0..3 -> 32-col strip
    const int m0 = blockIdx.y * 128;
    const int n0 = blockIdx.x * 128;
    const int l15 = lane & 15;
    const int lhi = lane >> 4;      // 0/1 half-wave

    v8f acc[4][2] = {};

    // Issue one LDS stage (128x64 A tile + 128x64 B tile) as 8 async B128
    // DMAs per thread; each wave instruction moves 512B memory->LDS.
    auto issue_stage = [&](int kt, int buf) {
        const int k0 = kt * 64;
#pragma unroll
        for (int i = 0; i < 4; ++i) {
            const int u   = tid + i * 256;     // 1024 x 16B chunks per tile
            const int row = u >> 3;
            const int c8  = (u & 7) * 8;       // bf16 offset inside 64-wide row
            async_load_b128_to_lds(
                (unsigned)(uintptr_t)&As[buf][row * LDSTRIDE + c8],
                &A[(size_t)(m0 + row) * K + k0 + c8]);
            async_load_b128_to_lds(
                (unsigned)(uintptr_t)&Bs[buf][row * LDSTRIDE + c8],
                &Bt[(size_t)(n0 + row) * K + k0 + c8]);
        }
    };

    const int ksteps = K / 64;
    issue_stage(0, 0);
    for (int kt = 0; kt < ksteps; ++kt) {
        const int buf = kt & 1;
        if (kt + 1 < ksteps) {
            issue_stage(kt + 1, buf ^ 1);      // overlap next stage with compute
            wait_asynccnt_le8();               // in-order: stage kt has landed
        } else {
            wait_asynccnt_0();
        }
        __syncthreads();                       // all waves' DMA portions visible

        // ---- two wmma k-steps of 32 ----
#pragma unroll
        for (int ks = 0; ks < 64; ks += 32) {
            v16bf afr[4];
#pragma unroll
            for (int fm = 0; fm < 4; ++fm) {
                // A lane layout: lane m holds K[0..7],K[16..23]; lane m+16: K[8..15],K[24..31]
                const bf16* p = &As[buf][(wm * 64 + fm * 16 + l15) * LDSTRIDE + ks + lhi * 8];
                ((uint4*)&afr[fm])[0] = *(const uint4*)p;
                ((uint4*)&afr[fm])[1] = *(const uint4*)(p + 16);
            }
            v16bf bfr[2];
#pragma unroll
            for (int fn = 0; fn < 2; ++fn) {
                // B lane layout: lane n holds K[0..7],K[16..23]; lane n+16: K[8..15],K[24..31]
                const bf16* p = &Bs[buf][(wn * 32 + fn * 16 + l15) * LDSTRIDE + ks + lhi * 8];
                ((uint4*)&bfr[fn])[0] = *(const uint4*)p;
                ((uint4*)&bfr[fn])[1] = *(const uint4*)(p + 16);
            }
#pragma unroll
            for (int fm = 0; fm < 4; ++fm)
#pragma unroll
                for (int fn = 0; fn < 2; ++fn)
                    acc[fm][fn] = __builtin_amdgcn_wmma_f32_16x16x32_bf16(
                        false, afr[fm], false, bfr[fn], (short)0, acc[fm][fn], false, false);
        }
        __syncthreads();   // readers done before this buffer is DMA-overwritten
    }

    // ---- epilogue: C layout lane(0-15)=n, vgpr r -> m = r (+8 for hi half) ----
#pragma unroll
    for (int fm = 0; fm < 4; ++fm)
#pragma unroll
        for (int fn = 0; fn < 2; ++fn)
#pragma unroll
            for (int r = 0; r < 8; ++r) {
                const int m = m0 + wm * 64 + fm * 16 + lhi * 8 + r;
                const int n = n0 + wn * 32 + fn * 16 + l15;
                float v = acc[fm][fn][r] + bias[n];
                if (EPI == 1 || EPI == 2) v = (v > 0.0f) ? (v + 1.0f) : __expf(v);
                if (EPI == 2 || EPI == 3) v *= rowmask[m];
                out[(size_t)m * N + n] = v;
            }
}

// ---------------- chunked scan, phase 1: per-chunk sums of k, k*v ----------
__global__ void chunk_sums(const float* __restrict__ K, const float* __restrict__ V,
                           float* __restrict__ sK, float* __restrict__ sKV) {
    const int blk = blockIdx.x;                 // B*H*NCHUNK
    const int c = blk % NCHUNK;
    const int h = (blk / NCHUNK) % DIM_H;
    const int b = blk / (NCHUNK * DIM_H);
    const int lane = threadIdx.x;               // 32 lanes, 4 channels each
    const int f = h * DIM_DH + lane * 4;
    float skx = 0, sky = 0, skz = 0, skw = 0;
    float svx = 0, svy = 0, svz = 0, svw = 0;
    for (int t = 0; t < CHUNK; ++t) {
        const size_t idx = ((size_t)b * DIM_L + (c * CHUNK + t)) * DIM_D + f;
        const float4 k4 = *(const float4*)(K + idx);
        const float4 v4 = *(const float4*)(V + idx);
        skx += k4.x; sky += k4.y; skz += k4.z; skw += k4.w;
        svx += k4.x * v4.x; svy += k4.y * v4.y; svz += k4.z * v4.z; svw += k4.w * v4.w;
    }
    const size_t o = (((size_t)(b * DIM_H + h) * NCHUNK + c) * DIM_DH) + lane * 4;
    *(float4*)(sK + o)  = make_float4(skx, sky, skz, skw);
    *(float4*)(sKV + o) = make_float4(svx, svy, svz, svw);
}

// ---------- phase 2: exclusive prefix over chunks, per (b,h,d) -------------
__global__ void chunk_prefix(float* __restrict__ sK, float* __restrict__ sKV) {
    const int bh = blockIdx.x;        // B*H
    const int d  = threadIdx.x;       // 128
    float ck = 0, ckv = 0;
    for (int c = 0; c < NCHUNK; ++c) {
        const size_t i = ((size_t)bh * NCHUNK + c) * DIM_DH + d;
        const float tk = sK[i];  sK[i]  = ck;  ck  += tk;
        const float tv = sKV[i]; sKV[i] = ckv; ckv += tv;
    }
}

// ---------- phase 3: local scan + carry, z-reduction, write att (bf16) -----
__global__ void scan_out(const float* __restrict__ Q, const float* __restrict__ K,
                         const float* __restrict__ V, const float* __restrict__ mask,
                         const float* __restrict__ cK, const float* __restrict__ cKV,
                         bf16* __restrict__ att) {
    const int blk = blockIdx.x;
    const int c = blk % NCHUNK;
    const int h = (blk / NCHUNK) % DIM_H;
    const int b = blk / (NCHUNK * DIM_H);
    const int lane = threadIdx.x;
    const int f = h * DIM_DH + lane * 4;
    const size_t o = (((size_t)(b * DIM_H + h) * NCHUNK + c) * DIM_DH) + lane * 4;
    float4 ck  = *(const float4*)(cK + o);
    float4 ckv = *(const float4*)(cKV + o);
    for (int t = 0; t < CHUNK; ++t) {
        const int l = c * CHUNK + t;
        const size_t idx = ((size_t)b * DIM_L + l) * DIM_D + f;
        const float4 q4 = *(const float4*)(Q + idx);
        const float4 k4 = *(const float4*)(K + idx);
        const float4 v4 = *(const float4*)(V + idx);
        ck.x += k4.x; ck.y += k4.y; ck.z += k4.z; ck.w += k4.w;
        ckv.x += k4.x * v4.x; ckv.y += k4.y * v4.y;
        ckv.z += k4.z * v4.z; ckv.w += k4.w * v4.w;
        float z = q4.x * ck.x + q4.y * ck.y + q4.z * ck.z + q4.w * ck.w;
#pragma unroll
        for (int off = 16; off > 0; off >>= 1)   // wave32 butterfly reduce
            z += __shfl_xor(z, off, 32);
        z = (z + 1e-6f) * mask[b * DIM_L + l];
        const float inv = 1.0f / z;
        union { bf16 hv[4]; uint2 u; } pk;
        pk.hv[0] = (bf16)(q4.x * ckv.x * inv);
        pk.hv[1] = (bf16)(q4.y * ckv.y * inv);
        pk.hv[2] = (bf16)(q4.z * ckv.z * inv);
        pk.hv[3] = (bf16)(q4.w * ckv.w * inv);
        *(uint2*)(att + idx) = pk.u;
    }
}

// ---------------------------------------------------------------------------
extern "C" void kernel_launch(void* const* d_in, const int* in_sizes, int n_in,
                              void* d_out, int out_size, void* d_ws, size_t ws_size,
                              hipStream_t stream) {
    (void)in_sizes; (void)n_in; (void)out_size; (void)ws_size;
    const float* x    = (const float*)d_in[0];
    const float* amask= (const float*)d_in[1];
    const float* Wq   = (const float*)d_in[2];
    const float* bq   = (const float*)d_in[3];
    const float* Wk   = (const float*)d_in[4];
    const float* bk   = (const float*)d_in[5];
    const float* Wv   = (const float*)d_in[6];
    const float* bv   = (const float*)d_in[7];
    const float* Wo   = (const float*)d_in[8];
    const float* bo   = (const float*)d_in[9];
    float* out = (float*)d_out;

    // workspace layout (bytes)
    char* ws = (char*)d_ws;
    const size_t SZ_XB  = (size_t)MTOK * DIM_D * 2;     // 64 MiB
    const size_t SZ_W   = (size_t)DIM_D * DIM_D * 2;    //  8 MiB each
    const size_t SZ_QKV = (size_t)MTOK * DIM_D * 4;     // 128 MiB each
    const size_t SZ_SUM = (size_t)DIM_B * DIM_H * NCHUNK * DIM_DH * 4;  // 2 MiB
    bf16*  xb   = (bf16*)(ws);
    bf16*  wqT  = (bf16*)(ws + SZ_XB);
    bf16*  wkT  = (bf16*)(ws + SZ_XB + SZ_W);
    bf16*  wvT  = (bf16*)(ws + SZ_XB + 2 * SZ_W);
    bf16*  woT  = (bf16*)(ws + SZ_XB + 3 * SZ_W);
    float* Qb   = (float*)(ws + SZ_XB + 4 * SZ_W);
    float* Kb   = (float*)(ws + SZ_XB + 4 * SZ_W + SZ_QKV);
    float* Vb   = (float*)(ws + SZ_XB + 4 * SZ_W + 2 * SZ_QKV);
    bf16*  attb = (bf16*) (ws + SZ_XB + 4 * SZ_W + 3 * SZ_QKV);
    float* sK   = (float*)(ws + 2 * SZ_XB + 4 * SZ_W + 3 * SZ_QKV);
    float* sKV  = (float*)(ws + 2 * SZ_XB + 4 * SZ_W + 3 * SZ_QKV + SZ_SUM);

    // 1) convert x to bf16; transpose+convert weights to bf16 [n][k]
    cvt_f32_bf16<<<8192, 256, 0, stream>>>(x, xb, MTOK * DIM_D);
    dim3 tg(DIM_D / 32, DIM_D / 32);
    transpose_cvt<<<tg, 256, 0, stream>>>(Wq, wqT, DIM_D);
    transpose_cvt<<<tg, 256, 0, stream>>>(Wk, wkT, DIM_D);
    transpose_cvt<<<tg, 256, 0, stream>>>(Wv, wvT, DIM_D);
    transpose_cvt<<<tg, 256, 0, stream>>>(Wo, woT, DIM_D);

    // 2) QKV projections with fused feature-map epilogues
    dim3 gg(DIM_D / 128, MTOK / 128);
    gemm_bf16<1><<<gg, 256, 0, stream>>>(xb, wqT, bq, amask, Qb, MTOK, DIM_D, DIM_D); // elu+1
    gemm_bf16<2><<<gg, 256, 0, stream>>>(xb, wkT, bk, amask, Kb, MTOK, DIM_D, DIM_D); // elu+1 * mask
    gemm_bf16<3><<<gg, 256, 0, stream>>>(xb, wvT, bv, amask, Vb, MTOK, DIM_D, DIM_D); // * mask

    // 3) chunked causal scan + normalization, write bf16 attention output
    chunk_sums <<<DIM_B * DIM_H * NCHUNK, 32, 0, stream>>>(Kb, Vb, sK, sKV);
    chunk_prefix<<<DIM_B * DIM_H, DIM_DH, 0, stream>>>(sK, sKV);
    scan_out   <<<DIM_B * DIM_H * NCHUNK, 32, 0, stream>>>(Qb, Kb, Vb, amask, sK, sKV, attb);

    // 4) output projection
    gemm_bf16<0><<<gg, 256, 0, stream>>>(attb, woT, bo, amask, out, MTOK, DIM_D, DIM_D);
}
